// TSE_59133109731926
// MI455X (gfx1250) — matrix operand order repaired and checked by reading, not compiled, not run on hardware
//
#include <hip/hip_runtime.h>
#include <hip/hip_bf16.h>
#include <math.h>

// ---------------------------------------------------------------------------
// MI455X / gfx1250 fused two-term attention (wave32, WMMA f32_16x16x32_f16)
//   K1: 5 projections (EW/ES @ W.T + b), f32 -> f16, LDS-staged weights
//   K2: E = exp(Q K^T + UQ UK^T) stored TRANSPOSED (E^T[j,i]) in f16 + row
//       sums S[b,i]; K/UK panels DMA'd into LDS by the Tensor Data Mover,
//       double-buffered, completion via s_wait_tensorcnt.
//   K3: out = (V_flat @ E) * (1/S[j])  -- V reinterpret makes A-rows contiguous
// ---------------------------------------------------------------------------

typedef __attribute__((ext_vector_type(16))) _Float16 v16h;
typedef __attribute__((ext_vector_type(8)))  _Float16 v8h;
typedef __attribute__((ext_vector_type(8)))  float    v8f;
typedef __attribute__((ext_vector_type(4)))  float    v4f;
typedef __attribute__((ext_vector_type(4)))  unsigned int u32x4;
typedef __attribute__((ext_vector_type(8)))  int          i32x8;
typedef __attribute__((ext_vector_type(4)))  int          i32x4;

#define DI __device__ __forceinline__

constexpr int Bn = 64, Ln = 512, Dn = 128;
constexpr float kScale = 0.04419417382415922f; // 1/(2*sqrt(128))

#if defined(__has_builtin)
#if __has_builtin(__builtin_amdgcn_tensor_load_to_lds) && \
    __has_builtin(__builtin_amdgcn_s_wait_tensorcnt)
#define HAVE_TDM 1
#endif
#endif
#ifndef HAVE_TDM
#define HAVE_TDM 0
#endif

DI v8f wmma16(v16h a, v16h b, v8f c) {
  // emits v_wmma_f32_16x16x32_f16
  return __builtin_amdgcn_wmma_f32_16x16x32_f16(
      false, a, false, b, (short)0, c, false, false);
}

// A fragment 16x32 f16 (row-major source). Per ISA layout: lane holds row
// m = lane&15; VGPRs 0-3 = K[half*8 .. +7], VGPRs 4-7 = K[16+half*8 .. +7].
DI v16h afrag_f16(const _Float16* __restrict__ rowk, int half) {
  const v8h lo = *(const v8h*)(rowk + half * 8);
  const v8h hi = *(const v8h*)(rowk + 16 + half * 8);
  v16h a;
#pragma unroll
  for (int t = 0; t < 8; ++t) { a[t] = lo[t]; a[t + 8] = hi[t]; }
  return a;
}

DI v16h afrag_f32(const float* __restrict__ rowk, int half) {
  v16h a;
#pragma unroll
  for (int t = 0; t < 8; ++t) {
    a[t]     = (_Float16)rowk[half * 8 + t];
    a[t + 8] = (_Float16)rowk[16 + half * 8 + t];
  }
  return a;
}

// B fragment 32x16 f16: lane holds column n = lane&15; elements 0-7 =
// K[half*8 .. +7], elements 8-15 = K[16+half*8 .. +7] of the column's
// source row (we always store B-sources row-per-column).
DI v16h bfrag_f16(const _Float16* __restrict__ colrow, int half) {
  const v8h lo = *(const v8h*)(colrow + half * 8);
  const v8h hi = *(const v8h*)(colrow + 16 + half * 8);
  v16h b;
#pragma unroll
  for (int t = 0; t < 8; ++t) { b[t] = lo[t]; b[t + 8] = hi[t]; }
  return b;
}

#if HAVE_TDM
// TDM: 1-D contiguous copy of n_f16 half-floats from global to LDS.
// D# per CDNA5 ISA:
//   g0: count=1 | lds_addr | global_addr[56:0] | type=2
//   g1: data_size=1(2B), tensor_dim0=n, tensor_dim1=1, tile_dim0=n,
//       tensor_dim0_stride=n; groups 2/3 zero (<=2D tensor).
DI void tdm_copy_f16(unsigned lds_byte_off, const _Float16* src, unsigned n_f16) {
  const unsigned long long ga = (unsigned long long)(uintptr_t)src;
  u32x4 g0;
  g0[0] = 1u;                                          // count=1, user D#
  g0[1] = lds_byte_off;                                // lds_addr
  g0[2] = (unsigned)(ga & 0xffffffffull);              // global_addr lo
  g0[3] = (unsigned)((ga >> 32) & 0x1ffffffull) | (2u << 30);  // hi | type=2
  i32x8 g1;
  g1[0] = (int)(1u << 16);                             // data_size = 2 bytes
  g1[1] = (int)((n_f16 & 0xffffu) << 16);              // tensor_dim0[15:0]
  g1[2] = (int)((n_f16 >> 16) | (1u << 16));           // dim0 hi | tensor_dim1=1
  g1[3] = (int)((n_f16 & 0xffffu) << 16);              // tile_dim0
  g1[4] = 0;
  g1[5] = (int)n_f16;                                  // tensor_dim0_stride lo
  g1[6] = 0;
  g1[7] = 0;
  const i32x4 z4 = {0, 0, 0, 0};
#if __clang_major__ >= 23
  const i32x8 z8 = {0, 0, 0, 0, 0, 0, 0, 0};
  __builtin_amdgcn_tensor_load_to_lds(g0, g1, z4, z4, z8, 0);
#else
  __builtin_amdgcn_tensor_load_to_lds(g0, g1, z4, z4, 0);
#endif
}
#endif

// ---------------------------------------------------------------------------
// Kernel 1: projections. grid = (B*L)/128 blocks x 256 thr (8 waves, 16 rows each)
// ---------------------------------------------------------------------------
__global__ __launch_bounds__(256) void k_proj(
    const float* __restrict__ EW,  const float* __restrict__ ES,
    const float* __restrict__ Wq,  const float* __restrict__ bq,
    const float* __restrict__ Wk,  const float* __restrict__ bk,
    const float* __restrict__ Wuq, const float* __restrict__ buq,
    const float* __restrict__ Wuk, const float* __restrict__ buk,
    const float* __restrict__ Wv,  const float* __restrict__ bv,
    _Float16* __restrict__ Qh, _Float16* __restrict__ Kh,
    _Float16* __restrict__ UQh, _Float16* __restrict__ UKh,
    _Float16* __restrict__ Vh) {
  __shared__ __align__(32) _Float16 wl[Dn * Dn];  // 32 KB staged weight (f16)
  __shared__ float bl[Dn];

  const int tid  = threadIdx.x;
  const int w    = tid >> 5;
  const int lane = tid & 31;
  const int half = lane >> 4;
  const int n16  = lane & 15;
  const int row0 = blockIdx.x * 128 + w * 16;

  // Preload A fragments for this wave's 16 rows of EW and ES (K=128 -> 4 steps)
  v16h aw[4], ae[4];
#pragma unroll
  for (int k = 0; k < 4; ++k) {
    const size_t r = (size_t)(row0 + n16) * Dn + k * 32;
    aw[k] = afrag_f32(EW + r, half);
    ae[k] = afrag_f32(ES + r, half);
  }

  const float* Ws[5]   = {Wq, Wk, Wuq, Wuk, Wv};
  const float* bs[5]   = {bq, bk, buq, buk, bv};
  _Float16*    dst[5]  = {Qh, Kh, UQh, UKh, Vh};
  const float  scl[5]  = {kScale, kScale, kScale, kScale, 1.0f};

#pragma unroll
  for (int p = 0; p < 5; ++p) {
    __syncthreads();
    const float* __restrict__ W = Ws[p];
    // Cooperative stage: 16384 f16 elems; each thread converts 8x8
    for (int t = tid; t < (Dn * Dn) / 8; t += 256) {
      const v4f w0 = *(const v4f*)(W + t * 8);
      const v4f w1 = *(const v4f*)(W + t * 8 + 4);
      v8h h;
#pragma unroll
      for (int e = 0; e < 4; ++e) { h[e] = (_Float16)w0[e]; h[e + 4] = (_Float16)w1[e]; }
      *(v8h*)(wl + t * 8) = h;
    }
    if (tid < Dn) bl[tid] = bs[p][tid];
    __syncthreads();

    const bool useES = (p == 2 || p == 3);
    _Float16* __restrict__ dp = dst[p];
#pragma unroll
    for (int nt = 0; nt < 8; ++nt) {
      v8f acc = {};
#pragma unroll
      for (int k = 0; k < 4; ++k) {
        const v16h a = useES ? ae[k] : aw[k];
        const v16h b = bfrag_f16(wl + (nt * 16 + n16) * Dn + k * 32, half);
        acc = wmma16(a, b, acc);
      }
      const int   col  = nt * 16 + n16;
      const float bias = bl[col];
#pragma unroll
      for (int v = 0; v < 8; ++v) {
        const int r = row0 + v + half * 8;
        dp[(size_t)r * Dn + col] = (_Float16)((acc[v] + bias) * scl[p]);
      }
    }
  }
}

// ---------------------------------------------------------------------------
// Kernel 2: E^T = exp(Q K^T + UQ UK^T)^T (f16) and row sums S[b,i] (f32).
// grid = B*4 blocks x 256 thr. Block: one batch, 128 rows; wave w: 16 rows,
// all 512 columns. K/UK 64-column panels (16 KB each, contiguous) are DMA'd
// into LDS by the Tensor Data Mover, double-buffered across panels.
// ---------------------------------------------------------------------------
__global__ __launch_bounds__(256) void k_scores(
    const _Float16* __restrict__ Qh,  const _Float16* __restrict__ Kh,
    const _Float16* __restrict__ UQh, const _Float16* __restrict__ UKh,
    _Float16* __restrict__ ET, float* __restrict__ S) {
  __shared__ __align__(32) _Float16 lk[2][64 * Dn];   // 2 x 16 KB
  __shared__ __align__(32) _Float16 luk[2][64 * Dn];  // 2 x 16 KB

  const int tid  = threadIdx.x;
  const int w    = tid >> 5;
  const int lane = tid & 31;
  const int half = lane >> 4;
  const int n16  = lane & 15;
  const int b    = blockIdx.x >> 2;
  const int i0   = (blockIdx.x & 3) * 128 + w * 16;

  const size_t rowbase = ((size_t)b * Ln + i0 + n16) * Dn;
  v16h aq[4], au[4];
#pragma unroll
  for (int k = 0; k < 4; ++k) {
    aq[k] = afrag_f16(Qh  + rowbase + k * 32, half);
    au[k] = afrag_f16(UQh + rowbase + k * 32, half);
  }

  float sums[8];
#pragma unroll
  for (int v = 0; v < 8; ++v) sums[v] = 0.0f;

  const size_t pbase = (size_t)b * Ln * Dn;  // element offset of this batch

#if HAVE_TDM
  // Prime the pipeline: panel 0 -> buffer 0 (wave 0 drives the TDM).
  if (tid < 32) {
    tdm_copy_f16((unsigned)(uintptr_t)&lk[0][0],  Kh  + pbase, 64 * Dn);
    tdm_copy_f16((unsigned)(uintptr_t)&luk[0][0], UKh + pbase, 64 * Dn);
  }
#endif

  for (int jc = 0; jc < 8; ++jc) {  // 8 panels of 64 columns
    const int buf = jc & 1;
#if HAVE_TDM
    if (tid < 32) {
      if (jc < 7) {  // prefetch next panel into the other buffer
        const size_t nsrow = pbase + (size_t)(jc + 1) * 64 * Dn;
        tdm_copy_f16((unsigned)(uintptr_t)&lk[buf ^ 1][0],  Kh  + nsrow, 64 * Dn);
        tdm_copy_f16((unsigned)(uintptr_t)&luk[buf ^ 1][0], UKh + nsrow, 64 * Dn);
        __builtin_amdgcn_s_wait_tensorcnt(2);  // current panel's 2 ops done
      } else {
        __builtin_amdgcn_s_wait_tensorcnt(0);
      }
    }
    __syncthreads();  // publish LDS panel to all waves
#else
    __syncthreads();
    const size_t srow = pbase + (size_t)jc * 64 * Dn;
    for (int t = tid; t < (64 * Dn) / 8; t += 256) {
      *(v8h*)(&lk[buf][0]  + t * 8) = *(const v8h*)(Kh  + srow + t * 8);
      *(v8h*)(&luk[buf][0] + t * 8) = *(const v8h*)(UKh + srow + t * 8);
    }
    __syncthreads();
#endif

#pragma unroll
    for (int jt = 0; jt < 4; ++jt) {
      v8f acc = {};
#pragma unroll
      for (int k = 0; k < 4; ++k) {
        acc = wmma16(aq[k], bfrag_f16(&lk[buf][0]  + (jt * 16 + n16) * Dn + k * 32, half), acc);
        acc = wmma16(au[k], bfrag_f16(&luk[buf][0] + (jt * 16 + n16) * Dn + k * 32, half), acc);
      }
      const int j = jc * 64 + jt * 16 + n16;
      _Float16* __restrict__ erow = ET + ((size_t)b * Ln + j) * Ln + i0;
      v8h eh;
#pragma unroll
      for (int v = 0; v < 8; ++v) {
        const float e = __expf(acc[v]);
        sums[v] += e;
        eh[v] = (_Float16)e;
      }
      *(v8h*)(erow + half * 8) = eh;  // contiguous in i -> one 16B store
    }
    __syncthreads();  // all waves done with buf before TDM overwrites it
  }

  // Reduce over the 16 lanes of each half (columns) -> full row sums
#pragma unroll
  for (int m = 1; m < 16; m <<= 1)
#pragma unroll
    for (int v = 0; v < 8; ++v) sums[v] += __shfl_xor(sums[v], m, 32);

  if (n16 < 8) {
    float sv = sums[0];
#pragma unroll
    for (int v = 1; v < 8; ++v) sv = (n16 == v) ? sums[v] : sv;
    S[(size_t)b * Ln + i0 + half * 8 + n16] = sv;
  }
}

// ---------------------------------------------------------------------------
// Kernel 3: out[b, p*512 + j] = (sum_l V_flat[b,p*512+l] * E[l,j]) / S[b,j]
// grid = B*8 blocks x 256 thr. Block: one batch, 64 columns; wave w: row tile w.
// ---------------------------------------------------------------------------
__global__ __launch_bounds__(256) void k_out(
    const _Float16* __restrict__ Vh, const _Float16* __restrict__ ET,
    const float* __restrict__ S, float* __restrict__ out) {
  const int tid  = threadIdx.x;
  const int w    = tid >> 5;
  const int lane = tid & 31;
  const int half = lane >> 4;
  const int n16  = lane & 15;
  const int b    = blockIdx.x >> 3;
  const int j0   = (blockIdx.x & 7) * 64;
  const int p0   = w * 16;

  const _Float16* __restrict__ Arow = Vh + (size_t)b * Ln * Dn + (size_t)(p0 + n16) * Ln;
  const size_t etb = (size_t)b * Ln * Ln;

  v8f acc[4] = {v8f{}, v8f{}, v8f{}, v8f{}};
  for (int k = 0; k < 16; ++k) {  // K = 512
    if (k < 15) __builtin_prefetch(Arow + (k + 1) * 32, 0, 1);
    const v16h a = afrag_f16(Arow + k * 32, half);
#pragma unroll
    for (int ct = 0; ct < 4; ++ct) {
      const int j = j0 + ct * 16 + n16;
      const v16h bf = bfrag_f16(ET + etb + (size_t)j * Ln + k * 32, half);
      acc[ct] = wmma16(a, bf, acc[ct]);
    }
  }

#pragma unroll
  for (int ct = 0; ct < 4; ++ct) {
    const int   j  = j0 + ct * 16 + n16;
    const float rs = 1.0f / S[(size_t)b * Ln + j];
    float* __restrict__ obase = out + (size_t)b * Ln * Dn + (size_t)p0 * Ln + j;
#pragma unroll
    for (int v = 0; v < 8; ++v)
      obase[(size_t)(v + half * 8) * Ln] = acc[ct][v] * rs;
  }
}

// ---------------------------------------------------------------------------
extern "C" void kernel_launch(void* const* d_in, const int* in_sizes, int n_in,
                              void* d_out, int out_size, void* d_ws, size_t ws_size,
                              hipStream_t stream) {
  const float* EW  = (const float*)d_in[0];
  const float* ES  = (const float*)d_in[1];
  const float* Wq  = (const float*)d_in[2];
  const float* bq  = (const float*)d_in[3];
  const float* Wk  = (const float*)d_in[4];
  const float* bk  = (const float*)d_in[5];
  const float* Wuq = (const float*)d_in[6];
  const float* buq = (const float*)d_in[7];
  const float* Wuk = (const float*)d_in[8];
  const float* buk = (const float*)d_in[9];
  const float* Wv  = (const float*)d_in[10];
  const float* bv  = (const float*)d_in[11];

  char* ws = (char*)d_ws;
  const size_t MB8 = 8ull << 20;  // B*L*D*2 bytes = 8 MiB exactly
  _Float16* Qh  = (_Float16*)(ws + 0 * MB8);
  _Float16* Kh  = (_Float16*)(ws + 1 * MB8);
  _Float16* UQh = (_Float16*)(ws + 2 * MB8);
  _Float16* UKh = (_Float16*)(ws + 3 * MB8);
  _Float16* Vh  = (_Float16*)(ws + 4 * MB8);
  _Float16* ET  = (_Float16*)(ws + 5 * MB8);                  // 32 MiB (B*L*L f16)
  float*    S   = (float*)   (ws + 5 * MB8 + (32ull << 20));  // 128 KiB

  k_proj<<<dim3((Bn * Ln) / 128), dim3(256), 0, stream>>>(
      EW, ES, Wq, bq, Wk, bk, Wuq, buq, Wuk, buk, Wv, bv,
      Qh, Kh, UQh, UKh, Vh);
  k_scores<<<dim3(Bn * 4), dim3(256), 0, stream>>>(Qh, Kh, UQh, UKh, ET, S);
  k_out<<<dim3(Bn * 8), dim3(256), 0, stream>>>(Vh, ET, S, (float*)d_out);
}
